// DirectedMessage_53412213293603
// MI455X (gfx1250) — matrix operand order repaired and checked by reading, not compiled, not run on hardware
//
#include <hip/hip_runtime.h>
#include <hip/hip_bf16.h>

typedef __attribute__((ext_vector_type(16))) __bf16 v16bf;
typedef __attribute__((ext_vector_type(8)))  __bf16 v8bf;
typedef __attribute__((ext_vector_type(8)))  float  v8f;

#define CAT   134      // CAT_DIM
#define NRBF  6
#define ADIM  42
#define NBIL  8
#define KPAD  160      // 5 * 32 (bf16 WMMA K step)
#define NPAD  144      // 9 * 16
#define MBLK  128      // angle rows per block
#define BT    256      // threads per block (8 wave32)
#define NT    9
#define KT    5

__device__ __forceinline__ __bf16 f2bf(float f) {
  // round-to-nearest-even float -> bf16 via bits (portable across toolchains)
  unsigned u = __float_as_uint(f);
  u += 0x7fffu + ((u >> 16) & 1u);
  unsigned short h = (unsigned short)(u >> 16);
  return __builtin_bit_cast(__bf16, h);
}

__global__ void zero_f32(float* p, int n) {
  int i = blockIdx.x * blockDim.x + threadIdx.x;
  if (i < n) p[i] = 0.f;
}

// W_m (134x134 f32, row = out channel) -> padded bf16 B-layout [NPAD][KPAD]
__global__ void conv_w(const float* __restrict__ Wm, __bf16* __restrict__ Wg) {
  int i = blockIdx.x * blockDim.x + threadIdx.x;
  if (i >= NPAD * KPAD) return;
  int n = i / KPAD, k = i - n * KPAD;
  float v = (n < CAT && k < CAT) ? Wm[n * CAT + k] : 0.f;
  Wg[i] = f2bf(v);
}

__launch_bounds__(BT)
__global__ void dimenet_interaction(
    const float* __restrict__ m_ji,
    const float* __restrict__ e_rbf,
    const float* __restrict__ a_sbf,
    const int*   __restrict__ kj_idx,
    const __bf16* __restrict__ Wg,       // [NPAD][KPAD] bf16, zero-padded
    const float* __restrict__ b_m,
    const float* __restrict__ W_e,       // [CAT][NRBF]
    const float* __restrict__ W_a,       // [NBIL][ADIM]
    const float* __restrict__ final_w,   // [NRBF][CAT]
    float* __restrict__ out,             // [E][NRBF], pre-zeroed
    int nAngles)
{
  __shared__ __align__(16) __bf16 Alds[MBLK * KPAD];  // gathered m_ji tile, bf16
  __shared__ float e6s[MBLK * NRBF];
  __shared__ float sas[MBLK];
  __shared__ float bms[NPAD];           // zero-padded -> unconditional loads
  __shared__ float wes[NPAD * NRBF];    // zero-padded
  __shared__ float fwt[NPAD * NRBF];    // final_w TRANSPOSED [ch][r], zero-padded
  __shared__ float cs[ADIM];
  __shared__ int   kjs[MBLK];

  const int tid = threadIdx.x;
  const int A0  = blockIdx.x * MBLK;

  // ---- stage 1: indices + small weights (padded to NPAD with zeros) ----
  if (tid < MBLK) {
    int a = A0 + tid;
    kjs[tid] = (a < nAngles) ? kj_idx[a] : -1;
  }
  if (tid < NPAD) bms[tid] = (tid < CAT) ? b_m[tid] : 0.f;
  for (int i = tid; i < NPAD * NRBF; i += BT) {
    int ch = i / NRBF, j = i - ch * NRBF;
    wes[i] = (ch < CAT) ? W_e[ch * NRBF + j] : 0.f;
    fwt[i] = (ch < CAT) ? final_w[j * CAT + ch] : 0.f;
  }
  if (tid < ADIM) {
    float s = 0.f;
    #pragma unroll
    for (int r = 0; r < NBIL; ++r) s += W_a[r * ADIM + tid];
    cs[tid] = s;   // colsum of W_a: sum(transf_a,-1) == a_sbf . cs
  }
  __syncthreads();

  // ---- stage 2: gather + convert tile data ----
  for (int i = tid; i < MBLK * KPAD; i += BT) {
    int row = i / KPAD, col = i - row * KPAD;
    int k = kjs[row];
    float v = (k >= 0 && col < CAT) ? m_ji[k * CAT + col] : 0.f;
    Alds[i] = f2bf(v);
  }
  for (int i = tid; i < MBLK * NRBF; i += BT) {
    int row = i / NRBF, j = i - row * NRBF;
    int k = kjs[row];
    e6s[i] = (k >= 0) ? e_rbf[k * NRBF + j] : 0.f;
  }
  if (tid < MBLK) {
    int a = A0 + tid;
    float s = 0.f;
    if (a < nAngles) {
      const float* ap = a_sbf + a * ADIM;
      #pragma unroll 6
      for (int j = 0; j < ADIM; ++j) s += ap[j] * cs[j];
    }
    sas[tid] = s;
  }
  __syncthreads();

  // ---- stage 3: WMMA GEMM (16 rows per wave x 9 N-tiles x 5 K-steps) ----
  const int wave = tid >> 5;         // row tile 0..7
  const int lane = tid & 31;
  const int lrow = lane & 15;
  const int hf   = lane >> 4;

  // rows this lane owns in the C layout: wave*16 + v + 8*hf
  float e6r[8][NRBF];
  #pragma unroll
  for (int v = 0; v < 8; ++v) {
    int lr = wave * 16 + v + 8 * hf;
    #pragma unroll
    for (int j = 0; j < NRBF; ++j) e6r[v][j] = e6s[lr * NRBF + j];
  }

  float part[8][NRBF];
  #pragma unroll
  for (int v = 0; v < 8; ++v)
    #pragma unroll
    for (int r = 0; r < NRBF; ++r) part[v][r] = 0.f;

  const int arow = wave * 16 + lrow;   // A-matrix row for this lane
  for (int nt = 0; nt < NT; ++nt) {
    const int ch = nt * 16 + lrow;     // output channel (C column == lane&15)
    v8f acc = {};
    #pragma unroll
    for (int kt = 0; kt < KT; ++kt) {
      union { v16bf v; v8bf h[2]; } afrag, bfrag;
      // A 16x32 bf16 layout: lane half selects K-group of 8; e0..7 -> K+0..7, e8..15 -> K+16..23
      const int abase = arow * KPAD + kt * 32 + 8 * hf;
      afrag.h[0] = *reinterpret_cast<const v8bf*>(&Alds[abase]);
      afrag.h[1] = *reinterpret_cast<const v8bf*>(&Alds[abase + 16]);
      // B 32x16 bf16: lane = N, same K order as A: e0..7 -> K+8*hf+0..7, e8..15 -> K+16+8*hf+0..7
      const __bf16* bp = Wg + ch * KPAD + kt * 32 + 8 * hf;
      bfrag.h[0] = *reinterpret_cast<const v8bf*>(bp);
      bfrag.h[1] = *reinterpret_cast<const v8bf*>(bp + 16);
      acc = __builtin_amdgcn_wmma_f32_16x16x32_bf16(
              false, afrag.v, false, bfrag.v, (short)0, acc, false, false);
    }
    // epilogue for this N-tile: all loads unconditional (padded LDS, zeros kill
    // contributions of channels >= CAT: acc=0, bmv=0 -> silu(0)=0; we=0 -> te=0)
    const float bmv = bms[ch];
    float we[NRBF], fw[NRBF];
    #pragma unroll
    for (int j = 0; j < NRBF; ++j) we[j] = wes[ch * NRBF + j];
    #pragma unroll
    for (int r = 0; r < NRBF; ++r) fw[r] = fwt[ch * NRBF + r];
    #pragma unroll
    for (int v = 0; v < 8; ++v) {
      float x  = acc[v] + bmv;
      float tm = x / (1.f + __expf(-x));   // silu
      float te = 0.f;
      #pragma unroll
      for (int j = 0; j < NRBF; ++j) te += we[j] * e6r[v][j];
      float me = tm * te;
      #pragma unroll
      for (int r = 0; r < NRBF; ++r) part[v][r] += me * fw[r];
    }
  }

  // ---- stage 4: reduce across the 16-lane half, scale by s_a, scatter-add ----
  #pragma unroll
  for (int v = 0; v < 8; ++v) {
    const int lr = wave * 16 + v + 8 * hf;
    const int a  = A0 + lr;
    const int k  = kjs[lr];        // LDS reads hoisted out of the guarded region
    const float sa = sas[lr];
    #pragma unroll
    for (int r = 0; r < NRBF; ++r) {
      float s = part[v][r];
      s += __shfl_xor(s, 1, 32);
      s += __shfl_xor(s, 2, 32);
      s += __shfl_xor(s, 4, 32);
      s += __shfl_xor(s, 8, 32);
      if (lrow == r && a < nAngles) {
        atomicAdd(&out[k * NRBF + r], s * sa);
      }
    }
  }
}

extern "C" void kernel_launch(void* const* d_in, const int* in_sizes, int n_in,
                              void* d_out, int out_size, void* d_ws, size_t ws_size,
                              hipStream_t stream) {
  const float* m_ji  = (const float*)d_in[0];
  // d_in[1] nbr_list, d_in[2] angle_list: unused by the reference
  const float* e_rbf = (const float*)d_in[3];
  const float* a_sbf = (const float*)d_in[4];
  const int*   kj    = (const int*)d_in[5];
  const float* W_m   = (const float*)d_in[6];
  const float* b_m   = (const float*)d_in[7];
  const float* W_e   = (const float*)d_in[8];
  const float* W_a   = (const float*)d_in[9];
  const float* fw    = (const float*)d_in[10];
  float*  out = (float*)d_out;
  __bf16* Wg  = (__bf16*)d_ws;          // 144*160*2 = 46080 B of scratch
  const int nAngles = in_sizes[5];

  zero_f32<<<(out_size + 255) / 256, 256, 0, stream>>>(out, out_size);
  conv_w<<<(NPAD * KPAD + 255) / 256, 256, 0, stream>>>(W_m, Wg);
  const int nblk = (nAngles + MBLK - 1) / MBLK;
  dimenet_interaction<<<nblk, BT, 0, stream>>>(m_ji, e_rbf, a_sbf, kj, Wg,
                                               b_m, W_e, W_a, fw, out, nAngles);
}
